// TemporalProposal_11321533792616
// MI455X (gfx1250) — matrix operand order repaired and checked by reading, not compiled, not run on hardware
//
#include <hip/hip_runtime.h>
#include <hip/hip_bf16.h>

// ---------------------------------------------------------------------------
// MI455X (gfx1250) implementation: all big GEMMs via v_wmma_f32_16x16x32_bf16
// (bf16 A/B, f32 accumulate). Wave32 throughout.
// Each wave computes a (16*MT) x (16*NT) tile: MT=2 halves B (weight) re-reads
// from L2 and raises the WMMA:load ratio to 8:12 per k-step.
// ---------------------------------------------------------------------------

typedef __bf16 bf16_t;
typedef bf16_t v16bf __attribute__((ext_vector_type(16)));
typedef bf16_t v8bf  __attribute__((ext_vector_type(8)));
typedef bf16_t v4bf  __attribute__((ext_vector_type(4)));
typedef float  v8f   __attribute__((ext_vector_type(8)));

constexpr int kT = 32768;
constexpr int kD = 1024;
constexpr int kC = 20;

enum { M_RELU_BF16 = 0, M_RELU_F32 = 1, M_FEAT = 2, M_TCAM = 3 };

__device__ __forceinline__ float sigmoidf(float x) {
  return 1.0f / (1.0f + __expf(-x));
}

// ---------------- f32 -> bf16 conversion (4 elems/thread) ------------------
__global__ void cvt_f32_bf16(const float* __restrict__ in, bf16_t* __restrict__ out, int n4) {
  int i = blockIdx.x * blockDim.x + threadIdx.x;
  if (i >= n4) return;
  float4 v = reinterpret_cast<const float4*>(in)[i];
  v4bf o = { (bf16_t)v.x, (bf16_t)v.y, (bf16_t)v.z, (bf16_t)v.w };
  reinterpret_cast<v4bf*>(out)[i] = o;
}

// ---------------- gated bf16 copy: out = bf16(feat * g1*g2*g3) -------------
// g1/g2/g3 are per-row scalars (any may be null). D fixed at 1024.
__global__ void gate_bf16_kernel(const float* __restrict__ feat,
                                 const float* __restrict__ g1,
                                 const float* __restrict__ g2,
                                 const float* __restrict__ g3,
                                 bf16_t* __restrict__ out) {
  int i = blockIdx.x * blockDim.x + threadIdx.x;   // index over T*D/4
  int t = (i * 4) >> 10;                           // D == 1024
  float g = 1.0f;
  if (g1) g *= g1[t];
  if (g2) g *= g2[t];
  if (g3) g *= g3[t];
  float4 v = reinterpret_cast<const float4*>(feat)[i];
  v4bf o = { (bf16_t)(v.x * g), (bf16_t)(v.y * g),
             (bf16_t)(v.z * g), (bf16_t)(v.w * g) };
  reinterpret_cast<v4bf*>(out)[i] = o;
}

// ---------------- erase mask from tcam + labels ----------------------------
__global__ void mask_kernel(const float* __restrict__ tcam, const int* __restrict__ labels,
                            float thresh, float* __restrict__ mask) {
  int t = blockIdx.x * blockDim.x + threadIdx.x;
  if (t >= kT) return;
  const float* row = tcam + (size_t)t * kC;
  int l0 = labels[0], l1 = labels[1], l2 = labels[2];
  float m = fmaxf(fmaxf(row[l0], row[l1]), row[l2]);
  mask[t] = (sigmoidf(m) >= thresh) ? 0.0f : 1.0f;
}

// ---------------- per-row dot (K=256) + sigmoid, one wave32 per row --------
__global__ void rowdot_sigmoid(const float* __restrict__ A, const float* __restrict__ w,
                               const float* __restrict__ bias, float* __restrict__ out,
                               int K, int rows) {
  int wave = blockIdx.x * (blockDim.x >> 5) + (threadIdx.x >> 5);
  int lane = threadIdx.x & 31;
  if (wave >= rows) return;
  const float* a = A + (size_t)wave * K;
  float s = 0.0f;
  for (int k = lane; k < K; k += 32) s += a[k] * w[k];
#pragma unroll
  for (int off = 16; off > 0; off >>= 1) s += __shfl_xor(s, off, 32);
  if (lane == 0) out[wave] = sigmoidf(s + (bias ? bias[0] : 0.0f));
}

__global__ void zero_f32(float* __restrict__ p, int n) {
  int i = blockIdx.x * blockDim.x + threadIdx.x;
  if (i < n) p[i] = 0.0f;
}

__global__ void finalize_x(const float* __restrict__ colsum,
                           float* __restrict__ x0, float* __restrict__ x1,
                           float* __restrict__ x2) {
  int i = threadIdx.x;
  if (i < kC) {
    x0[i] = sigmoidf(colsum[i]);
    x1[i] = sigmoidf(colsum[32 + i]);
    x2[i] = sigmoidf(colsum[64 + i]);
  }
}

// ---------------------------------------------------------------------------
// WMMA GEMM: Y[M,Nreal] = epilogue( X[M,K] @ W[Nreal,K]^T + bias )
// One wave computes a (16*MT) x (16*NT) tile, K-loop in steps of 32.
// A-fragment layout (16-bit A 16x32): lane half h holds contiguous K
// [16h..16h+15] (matching B) -> two 16B loads + concat.
// B-fragment: lane half h holds contiguous K [16h..16h+15] for column
// (lane&15) -> one 32B load. Each B fragment is shared by MT A fragments.
// ---------------------------------------------------------------------------
template <int MT, int NT, int MODE, bool HAS_BIAS, bool GUARD_N>
__global__ void gemm_wmma_bf16(const bf16_t* __restrict__ X,
                               const bf16_t* __restrict__ W,
                               const float*  __restrict__ bias,
                               int M, int Npad, int K, int Nreal,
                               float* __restrict__ outF,
                               bf16_t* __restrict__ outBf,
                               const float* __restrict__ resid,
                               float* __restrict__ colsum) {
  const int lane   = threadIdx.x & 31;
  const int waveId = blockIdx.x * (blockDim.x >> 5) + (threadIdx.x >> 5);
  const int colGroups = Npad / (16 * NT);
  const int rowTile  = waveId / colGroups;
  const int colGroup = waveId % colGroups;
  if (rowTile * 16 * MT >= M) return;  // uniform per wave: EXEC stays all-ones
  const int row0 = rowTile * 16 * MT;
  const int col0 = colGroup * 16 * NT;
  const int laneM  = lane & 15;
  const int laneHi = lane >> 4;

  v8f acc[MT][NT] = {};

  const bf16_t* aRow[MT];
#pragma unroll
  for (int mt = 0; mt < MT; ++mt)
    aRow[mt] = X + (size_t)(row0 + mt * 16 + laneM) * K + 16 * laneHi;
  const int wLaneOff = 16 * laneHi;

  for (int k0 = 0; k0 < K; k0 += 32) {
    v16bf a[MT];
#pragma unroll
    for (int mt = 0; mt < MT; ++mt) {
      v8bf alo = *reinterpret_cast<const v8bf*>(aRow[mt] + k0);
      v8bf ahi = *reinterpret_cast<const v8bf*>(aRow[mt] + k0 + 8);
      a[mt] = __builtin_shufflevector(alo, ahi,
          0, 1, 2, 3, 4, 5, 6, 7, 8, 9, 10, 11, 12, 13, 14, 15);
    }
#pragma unroll
    for (int nt = 0; nt < NT; ++nt) {
      const int col = col0 + nt * 16 + laneM;
      v16bf b = {};
      if (!GUARD_N || col < Nreal)
        b = *reinterpret_cast<const v16bf*>(W + (size_t)col * K + wLaneOff + k0);
#pragma unroll
      for (int mt = 0; mt < MT; ++mt)
        acc[mt][nt] = __builtin_amdgcn_wmma_f32_16x16x32_bf16(
            false, a[mt], false, b, (short)0, acc[mt][nt], false, false);
    }
  }

  // Epilogue. D-layout: VGPR v -> row (+ v + 8*laneHi), col (lane&15).
#pragma unroll
  for (int nt = 0; nt < NT; ++nt) {
    const int col = col0 + nt * 16 + laneM;
    const bool colOK = (!GUARD_N) || (col < Nreal);
    const float bv = (HAS_BIAS && colOK) ? bias[col] : 0.0f;
    float partial = 0.0f;
#pragma unroll
    for (int mt = 0; mt < MT; ++mt) {
#pragma unroll
      for (int v = 0; v < 8; ++v) {
        const int row = row0 + mt * 16 + v + 8 * laneHi;
        float val = acc[mt][nt][v] + bv;
        if (MODE == M_RELU_BF16) {
          if (colOK) outBf[(size_t)row * Nreal + col] = (bf16_t)fmaxf(val, 0.0f);
        } else if (MODE == M_RELU_F32) {
          if (colOK) outF[(size_t)row * Nreal + col] = fmaxf(val, 0.0f);
        } else if (MODE == M_FEAT) {
          val = fmaxf(val, 0.0f) + resid[(size_t)row * Nreal + col];
          outF[(size_t)row * Nreal + col]  = val;
          outBf[(size_t)row * Nreal + col] = (bf16_t)val;
        } else { // M_TCAM
          if (colOK) {
            outF[(size_t)row * Nreal + col] = val;
            partial += val;
          }
        }
      }
    }
    if (MODE == M_TCAM && colOK) atomicAdd(&colsum[col], partial);
  }
}

// ---------------------------------------------------------------------------
extern "C" void kernel_launch(void* const* d_in, const int* in_sizes, int n_in,
                              void* d_out_v, int out_size, void* d_ws, size_t ws_size,
                              hipStream_t stream) {
  (void)in_sizes; (void)n_in; (void)out_size; (void)ws_size;
  const float* inp     = (const float*)d_in[0];
  const int*   labels  = (const int*)d_in[1];
  const float* W_emb   = (const float*)d_in[2];
  const float* b_emb   = (const float*)d_in[3];
  const float* W_emb1  = (const float*)d_in[4];
  const float* b_emb1  = (const float*)d_in[5];
  const float* W_emb2  = (const float*)d_in[6];
  const float* b_emb2  = (const float*)d_in[7];
  const float* W_fc1   = (const float*)d_in[8];
  const float* b_fc1   = (const float*)d_in[9];
  const float* W_fc2   = (const float*)d_in[10];
  const float* b_fc2   = (const float*)d_in[11];
  const float* W_conv1 = (const float*)d_in[12];
  const float* b_conv1 = (const float*)d_in[13];
  const float* W_fc3   = (const float*)d_in[14];
  const float* W_fc4   = (const float*)d_in[15];
  const float* W_conv2 = (const float*)d_in[16];
  const float* W_fc5   = (const float*)d_in[17];
  const float* W_fc6   = (const float*)d_in[18];
  const float* W_conv3 = (const float*)d_in[19];

  // ---- output regions (flat, reference return order) ----
  float* out        = (float*)d_out_v;
  float* out_x      = out;                               // [20]
  float* out_feat   = out_x + kC;                        // [T,1024]
  float* out_tcam   = out_feat + (size_t)kT * kD;        // [T,20]
  float* out_ef1    = out_tcam + (size_t)kT * kC;        // [20]
  float* out_etcam  = out_ef1 + kC;                      // [T,20]
  float* out_eatt   = out_etcam + (size_t)kT * kC;       // [T]
  float* out_ef2    = out_eatt + kT;                     // [20]
  float* out_e2tcam = out_ef2 + kC;                      // [T,20]
  float* out_e2att  = out_e2tcam + (size_t)kT * kC;      // [T]

  // ---- workspace carve (~197 MB) ----
  char* base = (char*)d_ws;
  size_t off = 0;
  auto carve = [&](size_t bytes) -> void* {
    void* p = base + off;
    off = (off + bytes + 255) & ~(size_t)255;
    return p;
  };
  bf16_t* Wemb_bf   = (bf16_t*)carve((size_t)512 * 1024 * 2);
  bf16_t* Wemb1_bf  = (bf16_t*)carve((size_t)512 * 512 * 2);
  bf16_t* Wemb2_bf  = (bf16_t*)carve((size_t)1024 * 512 * 2);
  bf16_t* Wfc1_bf   = (bf16_t*)carve((size_t)256 * 1024 * 2);
  bf16_t* Wfc3_bf   = (bf16_t*)carve((size_t)256 * 1024 * 2);
  bf16_t* Wfc5_bf   = (bf16_t*)carve((size_t)256 * 1024 * 2);
  bf16_t* Wconv1_bf = (bf16_t*)carve((size_t)kC * 1024 * 2);
  bf16_t* Wconv2_bf = (bf16_t*)carve((size_t)kC * 1024 * 2);
  bf16_t* Wconv3_bf = (bf16_t*)carve((size_t)kC * 1024 * 2);
  bf16_t* bufA_bf   = (bf16_t*)carve((size_t)kT * kD * 2);   // Xbf, then featbf
  bf16_t* bufB_bf   = (bf16_t*)carve((size_t)kT * kD * 2);   // gated/erased copies
  bf16_t* H1_bf     = (bf16_t*)carve((size_t)kT * 512 * 2);  // also aliased as A1 f32
  bf16_t* H2_bf     = (bf16_t*)carve((size_t)kT * 512 * 2);
  float*  att1      = (float*)carve((size_t)kT * 4);
  float*  mask1     = (float*)carve((size_t)kT * 4);
  float*  mask2     = (float*)carve((size_t)kT * 4);
  float*  colsum    = (float*)carve(96 * 4);                 // 3 stages x 32 (20 used)
  float*  A1f       = (float*)H1_bf;                         // [T,256] f32 (same bytes)

  const int CB = 256;
  auto cvtGrid = [](size_t n) { return (int)((n / 4 + 255) / 256); };

  // ---- precision staging: f32 -> bf16 ----
  cvt_f32_bf16<<<cvtGrid((size_t)512 * 1024), CB, 0, stream>>>(W_emb,   Wemb_bf,   512 * 1024 / 4);
  cvt_f32_bf16<<<cvtGrid((size_t)512 * 512),  CB, 0, stream>>>(W_emb1,  Wemb1_bf,  512 * 512 / 4);
  cvt_f32_bf16<<<cvtGrid((size_t)1024 * 512), CB, 0, stream>>>(W_emb2,  Wemb2_bf,  1024 * 512 / 4);
  cvt_f32_bf16<<<cvtGrid((size_t)256 * 1024), CB, 0, stream>>>(W_fc1,   Wfc1_bf,   256 * 1024 / 4);
  cvt_f32_bf16<<<cvtGrid((size_t)256 * 1024), CB, 0, stream>>>(W_fc3,   Wfc3_bf,   256 * 1024 / 4);
  cvt_f32_bf16<<<cvtGrid((size_t)256 * 1024), CB, 0, stream>>>(W_fc5,   Wfc5_bf,   256 * 1024 / 4);
  cvt_f32_bf16<<<cvtGrid((size_t)kC * 1024),  CB, 0, stream>>>(W_conv1, Wconv1_bf, kC * 1024 / 4);
  cvt_f32_bf16<<<cvtGrid((size_t)kC * 1024),  CB, 0, stream>>>(W_conv2, Wconv2_bf, kC * 1024 / 4);
  cvt_f32_bf16<<<cvtGrid((size_t)kC * 1024),  CB, 0, stream>>>(W_conv3, Wconv3_bf, kC * 1024 / 4);
  cvt_f32_bf16<<<cvtGrid((size_t)kT * kD),    CB, 0, stream>>>(inp,     bufA_bf,   kT * kD / 4);
  zero_f32<<<1, 128, 0, stream>>>(colsum, 96);

  const int rowTiles32 = kT / 32;  // 1024 (MT = 2)
  const int gateGrid = kT * kD / (CB * 4);
  const int dotGrid  = kT / (CB / 32);
  const int maskGrid = kT / CB;

  // ---- embedding stack ----
  // H1 = relu(X @ Wemb^T + b_emb)      [T,512]
  gemm_wmma_bf16<2, 4, M_RELU_BF16, true, false>
      <<<rowTiles32 * (512 / 64) / 8, CB, 0, stream>>>(
      bufA_bf, Wemb_bf, b_emb, kT, 512, 1024, 512, nullptr, H1_bf, nullptr, nullptr);
  // H2 = relu(H1 @ Wemb1^T + b_emb1)   [T,512]
  gemm_wmma_bf16<2, 4, M_RELU_BF16, true, false>
      <<<rowTiles32 * (512 / 64) / 8, CB, 0, stream>>>(
      H1_bf, Wemb1_bf, b_emb1, kT, 512, 512, 512, nullptr, H2_bf, nullptr, nullptr);
  // feat = relu(H2 @ Wemb2^T + b_emb2) + inp   [T,1024], f32 out + bf16 copy
  gemm_wmma_bf16<2, 4, M_FEAT, true, false>
      <<<rowTiles32 * (1024 / 64) / 8, CB, 0, stream>>>(
      H2_bf, Wemb2_bf, b_emb2, kT, 1024, 512, 1024, out_feat, bufA_bf, inp, nullptr);

  // ---- stage 1: attention + tcam ----
  gemm_wmma_bf16<2, 4, M_RELU_F32, true, false>
      <<<rowTiles32 * (256 / 64) / 8, CB, 0, stream>>>(
      bufA_bf, Wfc1_bf, b_fc1, kT, 256, 1024, 256, A1f, nullptr, nullptr, nullptr);
  rowdot_sigmoid<<<dotGrid, CB, 0, stream>>>(A1f, W_fc2, b_fc2, att1, 256, kT);
  gate_bf16_kernel<<<gateGrid, CB, 0, stream>>>(out_feat, att1, nullptr, nullptr, bufB_bf);
  gemm_wmma_bf16<2, 2, M_TCAM, true, true>
      <<<rowTiles32 / 8, CB, 0, stream>>>(
      bufB_bf, Wconv1_bf, b_conv1, kT, 32, 1024, kC, out_tcam, nullptr, nullptr, colsum);

  // ---- erase stage 1 ----
  mask_kernel<<<maskGrid, CB, 0, stream>>>(out_tcam, labels, 0.53f, mask1);
  gate_bf16_kernel<<<gateGrid, CB, 0, stream>>>(out_feat, mask1, nullptr, nullptr, bufB_bf);
  gemm_wmma_bf16<2, 4, M_RELU_F32, false, false>
      <<<rowTiles32 * (256 / 64) / 8, CB, 0, stream>>>(
      bufB_bf, Wfc3_bf, nullptr, kT, 256, 1024, 256, A1f, nullptr, nullptr, nullptr);
  rowdot_sigmoid<<<dotGrid, CB, 0, stream>>>(A1f, W_fc4, nullptr, out_eatt, 256, kT);
  gate_bf16_kernel<<<gateGrid, CB, 0, stream>>>(out_feat, out_eatt, mask1, nullptr, bufB_bf);
  gemm_wmma_bf16<2, 2, M_TCAM, false, true>
      <<<rowTiles32 / 8, CB, 0, stream>>>(
      bufB_bf, Wconv2_bf, nullptr, kT, 32, 1024, kC, out_etcam, nullptr, nullptr, colsum + 32);

  // ---- erase stage 2 ----
  mask_kernel<<<maskGrid, CB, 0, stream>>>(out_etcam, labels, 0.51f, mask2);
  gate_bf16_kernel<<<gateGrid, CB, 0, stream>>>(out_feat, mask1, mask2, nullptr, bufB_bf);
  gemm_wmma_bf16<2, 4, M_RELU_F32, false, false>
      <<<rowTiles32 * (256 / 64) / 8, CB, 0, stream>>>(
      bufB_bf, Wfc5_bf, nullptr, kT, 256, 1024, 256, A1f, nullptr, nullptr, nullptr);
  rowdot_sigmoid<<<dotGrid, CB, 0, stream>>>(A1f, W_fc6, nullptr, out_e2att, 256, kT);
  gate_bf16_kernel<<<gateGrid, CB, 0, stream>>>(out_feat, out_e2att, mask1, mask2, bufB_bf);
  gemm_wmma_bf16<2, 2, M_TCAM, false, true>
      <<<rowTiles32 / 8, CB, 0, stream>>>(
      bufB_bf, Wconv3_bf, nullptr, kT, 32, 1024, kC, out_e2tcam, nullptr, nullptr, colsum + 64);

  // ---- final sigmoids over column sums ----
  finalize_x<<<1, 64, 0, stream>>>(colsum, out_x, out_ef1, out_ef2);
}
